// OmniCollectiveNetV4_44349832298632
// MI455X (gfx1250) — matrix operand order repaired and checked by reading, not compiled, not run on hardware
//
#include <hip/hip_runtime.h>
#include <hip/hip_bf16.h>
#include <cstdint>
#include <cstddef>

// ---------------------------------------------------------------------------
// CDNA5 / gfx1250 WMMA-based implementation of OmniCollectiveNetV4 forward.
// Dense math (linears + implicit-GEMM convs) -> v_wmma_f32_16x16x32_f16.
// Interior B panels are staged LDS-side with global_load_async_to_lds_b32
// (ASYNCcnt) per CDNA5 ISA; scalar kernels handle
// LN/softmax/topk/SE/GRU glue.
// ---------------------------------------------------------------------------

typedef __attribute__((ext_vector_type(16))) _Float16 v16h;
typedef __attribute__((ext_vector_type(8)))  _Float16 v8h;
typedef __attribute__((ext_vector_type(8)))  float    v8f;

#define ACT_NONE 0
#define ACT_SILU 1
#define ACT_SIGM 2

static inline int cdiv(int a, int b) { return (a + b - 1) / b; }

__device__ __forceinline__ float d_act(int act, float x) {
  if (act == ACT_SILU) return x / (1.f + __expf(-x));
  if (act == ACT_SIGM) return 1.f / (1.f + __expf(-x));
  return x;
}

// ------------------------------- accessors ---------------------------------
struct DenseA {
  const float* p; int K;
  static constexpr bool has_gp = true;
  __device__ __forceinline__ float ld(int r, int k) const {
    return (k < K) ? p[(long)r * K + k] : 0.f;
  }
  __device__ __forceinline__ const float* gp(int r, int k) const {
    return p + (long)r * K + k;
  }
};
struct Im2Col2dA {  // 3x3 SAME conv patches; K = C*9
  const float* x; int C, H, W, K, nchw;
  static constexpr bool has_gp = false;
  __device__ __forceinline__ float ld(int r, int k) const {
    if (k >= K) return 0.f;
    int ci = k / 9, t = k - ci * 9;
    int kh = t / 3 - 1, kw = (t % 3) - 1;
    int ow = r % W; int t2 = r / W; int oh = t2 % H; int b = t2 / H;
    int ih = oh + kh, iw = ow + kw;
    if ((unsigned)ih >= (unsigned)H || (unsigned)iw >= (unsigned)W) return 0.f;
    if (nchw) return x[(((long)b * C + ci) * H + ih) * W + iw];
    return x[(((long)b * H + ih) * W + iw) * C + ci];
  }
  __device__ __forceinline__ const float* gp(int, int) const { return x; }
};
struct Im2Col1dA {  // SAME 1d conv over T; x layout [B,T,C]; K = C*KK
  const float* x; int C, T, KK, K;
  static constexpr bool has_gp = false;
  __device__ __forceinline__ float ld(int r, int k) const {
    if (k >= K) return 0.f;
    int ci = k / KK, dt = k - ci * KK;
    int t = r % T, b = r / T;
    int it = t + dt - KK / 2;
    if ((unsigned)it >= (unsigned)T) return 0.f;
    return x[((long)b * T + it) * C + ci];
  }
  __device__ __forceinline__ const float* gp(int, int) const { return x; }
};
struct DenseB {  // weights [K,N] row-major (jax lin: (di,do))
  const float* p; int K, N;
  __device__ __forceinline__ float ld(int k, int n) const {
    return (k < K && n < N) ? p[(long)k * N + n] : 0.f;
  }
  __device__ __forceinline__ const float* gp(int k, int n) const {
    return p + (long)k * N + n;
  }
};
struct TransB {  // weights [N,K] row-major (conv OIHW / OIW flattened)
  const float* p; int K, N;
  __device__ __forceinline__ float ld(int k, int n) const {
    return (k < K && n < N) ? p[(long)n * K + k] : 0.f;
  }
  __device__ __forceinline__ const float* gp(int k, int n) const {
    return p + (long)n * K + k;
  }
};

// ------------------------------ WMMA GEMM ----------------------------------
// C[m, c0+n] (row stride ldC) = act( (A@B)[m,n]*cscale[n] + bias[n] )
// Block = 128 threads = 4 waves; block tile 16(M) x 64(N); K chunked by 32.
// A panel: f16 in LDS (converted while staging).  B panel: f32 in LDS; for
// fully-interior panels it is staged with per-lane async global->LDS copies
// (global_load_async_to_lds_b32, ASYNCcnt), else a guarded scalar fallback.
template <typename AA, typename BB>
__global__ void __launch_bounds__(128)
gemm_wmma(AA A, BB Bm, const float* __restrict__ bias,
          const float* __restrict__ cscale, float* __restrict__ C,
          int M, int N, int K, int act, int ldC, int c0) {
  __shared__ _Float16 As[16 * 32];
  __shared__ float Bs32[32 * 64];
  const int tid = threadIdx.x;
  const int wave = tid >> 5;
  const int lane = tid & 31;
  const int m0 = blockIdx.x * 16;
  const int n0 = blockIdx.y * 64;
  v8f acc = {};
  const int Kc = (K + 31) >> 5;
  for (int kc = 0; kc < Kc; ++kc) {
    const int kb = kc << 5;
    // ---- stage A (16x32 f16) ----
    if (AA::has_gp && (m0 + 16 <= M) && (kb + 32 <= K)) {
      // interior fast path: unguarded 4-element runs along K
      int r = tid >> 3, k4 = (tid & 7) * 4;
      const float* gp = A.gp(m0 + r, kb + k4);
#pragma unroll
      for (int u = 0; u < 4; ++u) As[r * 32 + k4 + u] = (_Float16)gp[u];
    } else {
      for (int i = tid; i < 16 * 32; i += 128) {
        int r = i >> 5, k = i & 31;
        float v = (m0 + r < M) ? A.ld(m0 + r, kb + k) : 0.f;
        As[i] = (_Float16)v;
      }
    }
    // ---- stage B (32x64 f32) ----
    if ((kb + 32 <= K) && (n0 + 64 <= N)) {
      // async global -> LDS copy, 16 x b32 per lane, tracked by ASYNCcnt
      for (int i = tid; i < 32 * 64; i += 128) {
        int k = i >> 6, n = i & 63;
        unsigned long long ga =
            (unsigned long long)(const void*)Bm.gp(kb + k, n0 + n);
        unsigned la = (unsigned)(uintptr_t)&Bs32[i];
        asm volatile("global_load_async_to_lds_b32 %0, %1, off"
                     :: "v"(la), "v"(ga) : "memory");
      }
      asm volatile("s_wait_asynccnt 0" ::: "memory");
    } else {
      for (int i = tid; i < 32 * 64; i += 128) {
        int k = i >> 6, n = i & 63;
        Bs32[i] = Bm.ld(kb + k, n0 + n);
      }
    }
    __syncthreads();
    // A fragment: 16-bit 16x32 layout (lanes 0-15: K 0..7 / 16..23;
    // lanes 16-31: K 8..15 / 24..31), row = lane&15. Two 16B LDS vectors.
    const int mr = lane & 15;
    const int sel = (lane < 16) ? 0 : 1;  // half-index within row
    const v8h* arow = (const v8h*)&As[mr * 32];
    v8h alo = arow[sel];
    v8h ahi = arow[sel + 2];
    v16h a = __builtin_shufflevector(alo, ahi, 0, 1, 2, 3, 4, 5, 6, 7, 8, 9,
                                     10, 11, 12, 13, 14, 15);
    // B fragment: column nc; lanes 0-15 hold K 0..7 / 16..23, lanes 16-31 K 8..15 / 24..31.
    const int nc = (wave << 4) + (lane & 15);
    const int kbb = (lane < 16) ? 0 : 8;
    v16h bfr;
#pragma unroll
    for (int e = 0; e < 16; ++e)
      bfr[e] = (_Float16)Bs32[(kbb + e + (e >= 8 ? 8 : 0)) * 64 + nc];
    acc = __builtin_amdgcn_wmma_f32_16x16x32_f16(false, a, false, bfr,
                                                 (short)0, acc, false, false);
    __syncthreads();
  }
  // C/D layout: lanes 0-15 rows m0+e, lanes 16-31 rows m0+8+e; col = lane&15.
  const int n = n0 + (wave << 4) + (lane & 15);
  if (n < N) {
    const float sc = cscale ? cscale[n] : 1.f;
    const float bi = bias ? bias[n] : 0.f;
    const int mbase = m0 + ((lane < 16) ? 0 : 8);
#pragma unroll
    for (int e = 0; e < 8; ++e) {
      int m = mbase + e;
      if (m < M) C[(long)m * ldC + c0 + n] = d_act(act, acc[e] * sc + bi);
    }
  }
}

// --------------------------- scalar helper kernels -------------------------
__global__ void k_zero(float* x, long n) {
  for (long i = blockIdx.x * (long)blockDim.x + threadIdx.x; i < n;
       i += (long)gridDim.x * blockDim.x) x[i] = 0.f;
}
__global__ void k_add(const float* x, const float* y, float* out, long n) {
  for (long i = blockIdx.x * (long)blockDim.x + threadIdx.x; i < n;
       i += (long)gridDim.x * blockDim.x) out[i] = x[i] + y[i];
}
__global__ void k_axpby(const float* x, const float* y, float* out, float a,
                        float b, long n) {
  for (long i = blockIdx.x * (long)blockDim.x + threadIdx.x; i < n;
       i += (long)gridDim.x * blockDim.x) out[i] = a * x[i] + b * y[i];
}
__global__ void k_mem_combine(float* fused, const float* gate,
                              const float* upd, const float* res, long n) {
  for (long i = blockIdx.x * (long)blockDim.x + threadIdx.x; i < n;
       i += (long)gridDim.x * blockDim.x)
    fused[i] = fused[i] + gate[i] * upd[i] + 0.12f * res[i];
}
__global__ void k_gate_combine(float* fused, const float* g, const float* v,
                               long n) {
  for (long i = blockIdx.x * (long)blockDim.x + threadIdx.x; i < n;
       i += (long)gridDim.x * blockDim.x) fused[i] += g[i] * v[i];
}
__global__ void k_row_scaled_add(float* dst, const float* src, const float* w,
                                 int col, int ncols, int F, long n) {
  for (long i = blockIdx.x * (long)blockDim.x + threadIdx.x; i < n;
       i += (long)gridDim.x * blockDim.x) {
    int b = (int)(i / F);
    dst[i] += w[b * ncols + col] * src[i];
  }
}
__global__ void k_copy2d(const float* src, float* dst, int M, int W,
                         int dstStride, int dstOff) {
  long n = (long)M * W;
  for (long i = blockIdx.x * (long)blockDim.x + threadIdx.x; i < n;
       i += (long)gridDim.x * blockDim.x) {
    int c = (int)(i % W);
    int r = (int)(i / W);
    dst[(long)r * dstStride + dstOff + c] = src[i];
  }
}
__global__ void k_gather_rows(const float* table, const int* ids, float* out,
                              long n_rows, int D) {
  for (long r = blockIdx.x; r < n_rows; r += gridDim.x)
    for (int j = threadIdx.x; j < D; j += blockDim.x)
      out[r * D + j] = table[(long)ids[r] * D + j];
}
__global__ void k_layernorm(const float* x, const float* g, const float* b,
                            float* y, int D) {
  int r = blockIdx.x;
  const float* xr = x + (long)r * D;
  __shared__ float s1[256], s2[256];
  float s = 0.f, q = 0.f;
  for (int j = threadIdx.x; j < D; j += blockDim.x) {
    float v = xr[j]; s += v; q += v * v;
  }
  s1[threadIdx.x] = s; s2[threadIdx.x] = q;
  __syncthreads();
  for (int off = 128; off > 0; off >>= 1) {
    if ((int)threadIdx.x < off) {
      s1[threadIdx.x] += s1[threadIdx.x + off];
      s2[threadIdx.x] += s2[threadIdx.x + off];
    }
    __syncthreads();
  }
  float mean = s1[0] / D;
  float var = s2[0] / D - mean * mean;
  float inv = rsqrtf(var + 1e-5f);
  for (int j = threadIdx.x; j < D; j += blockDim.x)
    y[(long)r * D + j] = (xr[j] - mean) * inv * g[j] + b[j];
}
__global__ void k_softmax_rows(float* x, int M, int D) {
  int r = blockIdx.x * blockDim.x + threadIdx.x;
  if (r >= M) return;
  float mx = -1e30f;
  for (int j = 0; j < D; ++j) mx = fmaxf(mx, x[(long)r * D + j]);
  float s = 0.f;
  for (int j = 0; j < D; ++j) {
    float e = __expf(x[(long)r * D + j] - mx);
    x[(long)r * D + j] = e; s += e;
  }
  for (int j = 0; j < D; ++j) x[(long)r * D + j] /= s;
}
__global__ void k_topk_norm(const float* p, float* w, int M, int D, int k) {
  int r = blockIdx.x * blockDim.x + threadIdx.x;
  if (r >= M) return;
  float tmp[16], out[16];
  for (int j = 0; j < D; ++j) { tmp[j] = p[(long)r * D + j]; out[j] = 0.f; }
  for (int i = 0; i < k; ++i) {
    int bi = 0; float bv = -1.f;
    for (int j = 0; j < D; ++j)
      if (tmp[j] > bv) { bv = tmp[j]; bi = j; }
    out[bi] = bv; tmp[bi] = -2.f;
  }
  float s = 0.f;
  for (int j = 0; j < D; ++j) s += out[j];
  s = fmaxf(s, 1e-6f);
  for (int j = 0; j < D; ++j) w[(long)r * D + j] = out[j] / s;
}
__global__ void k_loss_accum(const float* probs, int M, int D,
                             float* loss_slot) {
  __shared__ float cm[16];
  if ((int)threadIdx.x < D) {
    float s = 0.f;
    for (int r = 0; r < M; ++r) s += probs[(long)r * D + threadIdx.x];
    cm[threadIdx.x] = s / M;
  }
  __syncthreads();
  if (threadIdx.x == 0) {
    float t = 0.f;
    for (int j = 0; j < D; ++j) {
      float d = cm[j] - 1.f / D; t += d * d;
    }
    *loss_slot += t / D;
  }
}
__global__ void k_ctx_sum(const float* stack, const float* cw, float* out,
                          int Mo, int F) {
  int b = blockIdx.x;
  for (int f = threadIdx.x; f < F; f += blockDim.x) {
    float s = 0.f;
    for (int m = 0; m < Mo; ++m)
      s += cw[b * Mo + m] * stack[((long)b * Mo + m) * F + f];
    out[(long)b * F + f] = s;
  }
}
__global__ void k_maxpool2_nhwc(const float* x, float* y, int B, int H, int W,
                                int C) {
  int Ho = H / 2, Wo = W / 2;
  long n = (long)B * Ho * Wo * C;
  for (long i = blockIdx.x * (long)blockDim.x + threadIdx.x; i < n;
       i += (long)gridDim.x * blockDim.x) {
    int c = (int)(i % C); long t = i / C;
    int ow = (int)(t % Wo); t /= Wo;
    int oh = (int)(t % Ho); int b = (int)(t / Ho);
    const float* p = x + (((long)b * H + 2 * oh) * W + 2 * ow) * C + c;
    float v = fmaxf(fmaxf(p[0], p[C]),
                    fmaxf(p[(long)W * C], p[(long)W * C + C]));
    y[i] = v;
  }
}
// x[(b*L+l)*C+c] -> out[b*C+c]; op 0 = mean, 1 = max. grid = B*C blocks.
__global__ void k_reduce_mid(const float* x, float* out, int L, int C, int op) {
  int bc = blockIdx.x;
  int b = bc / C, c = bc % C;
  __shared__ float red[256];
  float acc = (op == 1) ? -1e30f : 0.f;
  for (int l = threadIdx.x; l < L; l += blockDim.x) {
    float v = x[((long)b * L + l) * C + c];
    acc = (op == 1) ? fmaxf(acc, v) : acc + v;
  }
  red[threadIdx.x] = acc;
  __syncthreads();
  for (int off = 128; off > 0; off >>= 1) {
    if ((int)threadIdx.x < off)
      red[threadIdx.x] = (op == 1) ? fmaxf(red[threadIdx.x],
                                           red[threadIdx.x + off])
                                   : red[threadIdx.x] + red[threadIdx.x + off];
    __syncthreads();
  }
  if (threadIdx.x == 0) out[bc] = (op == 1) ? red[0] : red[0] / L;
}
__global__ void k_scale_channels(float* x, const float* s, long HW, int C,
                                 long n) {
  for (long i = blockIdx.x * (long)blockDim.x + threadIdx.x; i < n;
       i += (long)gridDim.x * blockDim.x) {
    int c = (int)(i % C);
    int b = (int)((i / C) / HW);
    x[i] *= s[(long)b * C + c];
  }
}
__global__ void k_word_mean(const float* wemb, const int* ids, float* out,
                            int B, int Wn, int D) {
  int b = blockIdx.x;
  __shared__ float cnt;
  if (threadIdx.x == 0) {
    int c = 0;
    for (int w = 0; w < Wn; ++w)
      if (ids[b * Wn + w] != 0) c++;
    cnt = fmaxf((float)c, 1.f);
  }
  __syncthreads();
  for (int d = threadIdx.x; d < D; d += blockDim.x) {
    float s = 0.f;
    for (int w = 0; w < Wn; ++w) {
      int id = ids[b * Wn + w];
      if (id != 0) s += wemb[(long)id * D + d];
    }
    out[(long)b * D + d] = s / cnt;
  }
}
// GRU recurrence: one block, h kept in LDS. gi rows = b*T+t, layout [.,3H].
__global__ void __launch_bounds__(1024)
k_gru_seq(const float* gi, const float* whh, const float* bhh, float* out_cat,
          int outStride, int outOff, int T, int B, int H, int reverse) {
  __shared__ float h[64 * 160];
  const int BH = B * H;
  for (int i = threadIdx.x; i < BH; i += blockDim.x) h[i] = 0.f;
  __syncthreads();
  float newh[12];
  for (int s = 0; s < T; ++s) {
    int t = reverse ? (T - 1 - s) : s;
    int cnt = 0;
    for (int i = threadIdx.x; i < BH; i += blockDim.x) {
      int b = i / H, j = i - b * H;
      const float* gir = gi + ((long)b * T + t) * (3 * H);
      float hr = bhh[j], hz = bhh[H + j], hn = bhh[2 * H + j];
      const float* hb = h + b * H;
      for (int k2 = 0; k2 < H; ++k2) {
        float hk = hb[k2];
        const float* wr = whh + (long)k2 * 3 * H;
        hr += hk * wr[j];
        hz += hk * wr[H + j];
        hn += hk * wr[2 * H + j];
      }
      float r = 1.f / (1.f + __expf(-(gir[j] + hr)));
      float z = 1.f / (1.f + __expf(-(gir[H + j] + hz)));
      float n = tanhf(gir[2 * H + j] + r * hn);
      if (cnt < 12) newh[cnt] = (1.f - z) * n + z * hb[j];
      cnt++;
    }
    __syncthreads();
    cnt = 0;
    for (int i = threadIdx.x; i < BH; i += blockDim.x) {
      if (cnt < 12) h[i] = newh[cnt];
      cnt++;
    }
    __syncthreads();
  }
  for (int i = threadIdx.x; i < BH; i += blockDim.x) {
    int b = i / H, j = i - b * H;
    out_cat[(long)b * outStride + outOff + j] = h[i];
  }
}

// ----------------------------- host wrappers -------------------------------
static inline dim3 ew_grid(long n) {
  long b = (n + 255) / 256;
  if (b > 524288) b = 524288;
  return dim3((unsigned)b);
}
template <typename AA, typename BB>
static void launch_gemm(hipStream_t s, AA a, BB b, const float* bias,
                        const float* cs, float* C, int M, int N, int K, int act,
                        int ldC, int c0) {
  dim3 g(cdiv(M, 16), cdiv(N, 64));
  gemm_wmma<AA, BB><<<g, dim3(128), 0, s>>>(a, b, bias, cs, C, M, N, K, act,
                                            ldC, c0);
}
static void gemm_lin(hipStream_t s, const float* A, const float* W,
                     const float* bias, float* C, int M, int N, int K, int act,
                     int ldC = -1, int c0 = 0) {
  DenseA a{A, K}; DenseB b{W, K, N};
  launch_gemm(s, a, b, bias, (const float*)nullptr, C, M, N, K, act,
              ldC < 0 ? N : ldC, c0);
}
static void conv2d_bn(hipStream_t s, const float* x, int B, int C, int H,
                      int W, int nchw, const float* wOIHW, int Cout,
                      const float* g, const float* bb, float* y, int act) {
  int K = C * 9, M = B * H * W;
  Im2Col2dA a{x, C, H, W, K, nchw};
  TransB b{wOIHW, K, Cout};
  launch_gemm(s, a, b, bb, g, y, M, Cout, K, act, Cout, 0);
}
static void conv1d_b(hipStream_t s, const float* x, int B, int C, int T,
                     int KK, const float* w, int Cout, const float* bias,
                     float* y, int act) {
  int K = C * KK, M = B * T;
  Im2Col1dA a{x, C, T, KK, K};
  TransB b{w, K, Cout};
  launch_gemm(s, a, b, bias, (const float*)nullptr, y, M, Cout, K, act, Cout,
              0);
}

// --------------------------------- driver ----------------------------------
extern "C" void kernel_launch(void* const* d_in, const int* in_sizes, int n_in,
                              void* d_out, int out_size, void* d_ws,
                              size_t ws_size, hipStream_t stream) {
  (void)in_sizes; (void)n_in; (void)out_size; (void)ws_size;
  const int B = 64, T = 256, E = 84, H = 160, G3 = 480, F = 576;
  const int* token_ids = (const int*)d_in[0];
  const float* image = (const float*)d_in[1];
  const float* has_image = (const float*)d_in[2];
  const int* word_ids = (const int*)d_in[3];
  const float* prompt = (const float*)d_in[4];
#define PF(i) ((const float*)d_in[i])
  // jax pytree flatten order (dict keys sorted) of params, after 5 inputs:
  enum {
    P_CC1_B = 5, P_CC1_W, P_CC2_B, P_CC2_W, P_CC3_B, P_CC3_W,
    P_CHNORM_B, P_CHNORM_G, P_CHPROJ_B, P_CHPROJ_W,
    P_CQ0_B, P_CQ0_W, P_CQ1_B, P_CQ1_W, P_CQ2_B, P_CQ2_W, P_CQ3_B, P_CQ3_W,
    P_CQ4_B, P_CQ4_W,
    P_DINORM_B, P_DINORM_G,
    P_D_BN1_B, P_D_BN1_G, P_D_BN2_B, P_D_BN2_G, P_D_BN3_B, P_D_BN3_G,
    P_D_BN4_B, P_D_BN4_G,
    P_D_C1, P_D_C2, P_D_C3, P_D_C4,
    P_D_SE1_F1B, P_D_SE1_F1W, P_D_SE1_F2B, P_D_SE1_F2W,
    P_D_SE2_F1B, P_D_SE2_F1W, P_D_SE2_F2B, P_D_SE2_F2W,
    P_DIMGPROJ_B, P_DIMGPROJ_W,
    P_DOMHEAD_B, P_DOMHEAD_W,
    P_EMB,
    P_EINNORM_B, P_EINNORM_G,
    P_EMERGE_B, P_EMERGE_W,
    P_ER0_B, P_ER0_W, P_ER1_B, P_ER1_W, P_ER2_B, P_ER2_W, P_ER3_B, P_ER3_W,
    P_ER4_B, P_ER4_W,
    P_EX0_F1B, P_EX0_F1W, P_EX0_F2B, P_EX0_F2W, P_EX0_LNB, P_EX0_LNG,
    P_EX1_F1B, P_EX1_F1W, P_EX1_F2B, P_EX1_F2W, P_EX1_LNB, P_EX1_LNG,
    P_EX2_F1B, P_EX2_F1W, P_EX2_F2B, P_EX2_F2W, P_EX2_LNB, P_EX2_LNG,
    P_EX3_F1B, P_EX3_F1W, P_EX3_F2B, P_EX3_F2W, P_EX3_LNB, P_EX3_LNG,
    P_FUSIN_B, P_FUSIN_W,
    P_GRUB_BHH, P_GRUB_BIH, P_GRUB_WHH, P_GRUB_WIH,
    P_GRUF_BHH, P_GRUF_BIH, P_GRUF_WHH, P_GRUF_WIH,
    P_HIPROJ_B, P_HIPROJ_W,
    P_IMNORM_B, P_IMNORM_G, P_IMPROJ_B, P_IMPROJ_W,
    P_I_BN1_B, P_I_BN1_G, P_I_BN2_B, P_I_BN2_G, P_I_BN3_B, P_I_BN3_G,
    P_I_BN4_B, P_I_BN4_G,
    P_I_C1, P_I_C2, P_I_C3, P_I_C4,
    P_I_SE_F1B, P_I_SE_F1W, P_I_SE_F2B, P_I_SE_F2W,
    P_INTHEAD_B, P_INTHEAD_W,
    P_MEM_GATE_B, P_MEM_GATE_W, P_MEM_MIX1_B, P_MEM_MIX1_W,
    P_MEM_MIX2_B, P_MEM_MIX2_W, P_MEM_MLN_B, P_MEM_MLN_G,
    P_MEM_Q_B, P_MEM_Q_W, P_MEM_RES_B, P_MEM_RES_W, P_MEM_SLOTS,
    P_PF1_B, P_PF1_W, P_PF2_B, P_PF2_W,
    P_PREDOM_B, P_PREDOM_W, P_PREINT_B, P_PREINT_W, P_PREVIS_B, P_PREVIS_W,
    P_PRNORM_B, P_PRNORM_G, P_PRPROJ_B, P_PRPROJ_W,
    P_REF_F1B, P_REF_F1W, P_REF_F2B, P_REF_F2W, P_REF_LNB, P_REF_LNG,
    P_RESPHEAD_B, P_RESPHEAD_W,
    P_TNORM_B, P_TNORM_G, P_TPROJ_B, P_TPROJ_W,
    P_VGATE_B, P_VGATE_W,
    P_VISHEAD_B, P_VISHEAD_W,
    P_VP1_B, P_VP1_W, P_VP2_B, P_VP2_W,
    P_WEMB,
    P_WNORM_B, P_WNORM_G, P_WPROJ_B, P_WPROJ_W
  };

  // ---- workspace bump allocator -------------------------------------------
  size_t cur = 0;
  auto alloc = [&](size_t nf) -> float* {
    float* p = (float*)((char*)d_ws + cur);
    cur += ((nf * sizeof(float) + 255) / 256) * 256;
    return p;
  };
  float* emb = alloc((size_t)B * T * E);
  float* gi_f = alloc((size_t)B * T * G3);
  float* gi_b = alloc((size_t)B * T * G3);
  float* text_cat = alloc((size_t)B * 320);
  float* text_state = alloc((size_t)B * 320);
  float* chA = alloc((size_t)B * T * 192);
  float* chB = alloc((size_t)B * T * 192);
  float* char_state = alloc((size_t)B * 192);
  float* word_state = alloc((size_t)B * 72);
  float* ptmp = alloc((size_t)B * 64);
  float* ptmp2 = alloc((size_t)B * 40);
  float* prompt_state = alloc((size_t)B * 40);
  float* feat = alloc((size_t)B * 192);
  float* image_state = alloc((size_t)B * 108);
  float* dimg_state = alloc((size_t)B * 192);
  float* se_mean = alloc((size_t)B * 192);
  float* se_hid = alloc((size_t)B * 64);
  float* se_scale = alloc((size_t)B * 192);
  float* fin = alloc((size_t)B * 925);
  float* fused = alloc((size_t)B * F);
  float* stack = alloc((size_t)B * 7 * F);
  float* attn = alloc((size_t)B * 12);
  float* memv = alloc((size_t)B * F);
  float* joined = alloc((size_t)B * 2 * F);
  float* jln = alloc((size_t)B * 2 * F);
  float* mixh = alloc((size_t)B * 2 * F);
  float* upd = alloc((size_t)B * F);
  float* gatev = alloc((size_t)B * F);
  float* resid = alloc((size_t)B * F);
  float* cl = alloc((size_t)B * 7);
  float* cw = alloc((size_t)B * 7);
  float* ctx = alloc((size_t)B * F);
  float* tmpF = alloc((size_t)B * F);
  float* routed = alloc((size_t)B * F);
  float* rcat = alloc((size_t)B * 2 * F);
  float* rl = alloc((size_t)B * 4);
  float* rw = alloc((size_t)B * 4);
  float* ein = alloc((size_t)B * F);
  float* mixed = alloc((size_t)B * F);
  float* eh = alloc((size_t)B * F);
  float* e1 = alloc((size_t)B * 896);
  float* e2 = alloc((size_t)B * F);
  float* eout = alloc((size_t)B * F);
  float* ccat = alloc((size_t)B * 2 * F);
  float* delta = alloc((size_t)B * F);
  float* pre3 = alloc((size_t)B * 260);
  float* vv1 = alloc((size_t)B * 288);
  float* verifier = alloc((size_t)B * F);
  float* gcat = alloc((size_t)B * 2 * F);
  float* vg = alloc((size_t)B * F);
  float* refined = alloc((size_t)B * F);
  float* bigA = alloc((size_t)64 * 160 * 160 * 64);  // 419 MB conv ping
  float* bigB = alloc((size_t)64 * 80 * 80 * 64);    // 105 MB conv pong

  float* out = (float*)d_out;
  float* ctx_slot = out + 82176;
  float* exp_slot = out + 82177;
  k_zero<<<1, 64, 0, stream>>>(ctx_slot, 2);

  hipStream_t s = stream;
  // ---- token embedding + GRU ---------------------------------------------
  k_gather_rows<<<dim3(4096), dim3(128), 0, s>>>(PF(P_EMB), token_ids, emb,
                                                 (long)B * T, E);
  gemm_lin(s, emb, PF(P_GRUF_WIH), PF(P_GRUF_BIH), gi_f, B * T, G3, E,
           ACT_NONE);
  gemm_lin(s, emb, PF(P_GRUB_WIH), PF(P_GRUB_BIH), gi_b, B * T, G3, E,
           ACT_NONE);
  k_gru_seq<<<1, 1024, 0, s>>>(gi_f, PF(P_GRUF_WHH), PF(P_GRUF_BHH), text_cat,
                               320, 0, T, B, H, 0);
  k_gru_seq<<<1, 1024, 0, s>>>(gi_b, PF(P_GRUB_WHH), PF(P_GRUB_BHH), text_cat,
                               320, 160, T, B, H, 1);
  k_layernorm<<<B, 256, 0, s>>>(text_cat, PF(P_TNORM_G), PF(P_TNORM_B),
                                text_state, 320);
  // ---- image branch (NHWC after first conv) ------------------------------
  conv2d_bn(s, image, B, 3, 160, 160, 1, PF(P_I_C1), 36, PF(P_I_BN1_G),
            PF(P_I_BN1_B), bigA, ACT_SILU);
  k_maxpool2_nhwc<<<ew_grid((long)B * 80 * 80 * 36), 256, 0, s>>>(bigA, bigB,
                                                                  B, 160, 160,
                                                                  36);
  conv2d_bn(s, bigB, B, 36, 80, 80, 0, PF(P_I_C2), 72, PF(P_I_BN2_G),
            PF(P_I_BN2_B), bigA, ACT_SILU);
  k_maxpool2_nhwc<<<ew_grid((long)B * 40 * 40 * 72), 256, 0, s>>>(bigA, bigB,
                                                                  B, 80, 80,
                                                                  72);
  conv2d_bn(s, bigB, B, 72, 40, 40, 0, PF(P_I_C3), 108, PF(P_I_BN3_G),
            PF(P_I_BN3_B), bigA, ACT_SILU);
  k_reduce_mid<<<B * 108, 256, 0, s>>>(bigA, se_mean, 1600, 108, 0);
  gemm_lin(s, se_mean, PF(P_I_SE_F1W), PF(P_I_SE_F1B), se_hid, B, 27, 108,
           ACT_SILU);
  gemm_lin(s, se_hid, PF(P_I_SE_F2W), PF(P_I_SE_F2B), se_scale, B, 108, 27,
           ACT_SIGM);
  k_scale_channels<<<ew_grid((long)B * 1600 * 108), 256, 0, s>>>(
      bigA, se_scale, 1600, 108, (long)B * 1600 * 108);
  conv2d_bn(s, bigA, B, 108, 40, 40, 0, PF(P_I_C4), 108, PF(P_I_BN4_G),
            PF(P_I_BN4_B), bigB, ACT_SILU);
  k_reduce_mid<<<B * 108, 256, 0, s>>>(bigB, feat, 1600, 108, 0);
  k_layernorm<<<B, 256, 0, s>>>(feat, PF(P_IMNORM_G), PF(P_IMNORM_B),
                                image_state, 108);
  // ---- deep image branch --------------------------------------------------
  conv2d_bn(s, image, B, 3, 160, 160, 1, PF(P_D_C1), 64, PF(P_D_BN1_G),
            PF(P_D_BN1_B), bigA, ACT_SILU);
  k_maxpool2_nhwc<<<ew_grid((long)B * 80 * 80 * 64), 256, 0, s>>>(bigA, bigB,
                                                                  B, 160, 160,
                                                                  64);
  conv2d_bn(s, bigB, B, 64, 80, 80, 0, PF(P_D_C2), 128, PF(P_D_BN2_G),
            PF(P_D_BN2_B), bigA, ACT_SILU);
  k_reduce_mid<<<B * 128, 256, 0, s>>>(bigA, se_mean, 6400, 128, 0);
  gemm_lin(s, se_mean, PF(P_D_SE1_F1W), PF(P_D_SE1_F1B), se_hid, B, 32, 128,
           ACT_SILU);
  gemm_lin(s, se_hid, PF(P_D_SE1_F2W), PF(P_D_SE1_F2B), se_scale, B, 128, 32,
           ACT_SIGM);
  k_scale_channels<<<ew_grid((long)B * 6400 * 128), 256, 0, s>>>(
      bigA, se_scale, 6400, 128, (long)B * 6400 * 128);
  k_maxpool2_nhwc<<<ew_grid((long)B * 40 * 40 * 128), 256, 0, s>>>(
      bigA, bigB, B, 80, 80, 128);
  conv2d_bn(s, bigB, B, 128, 40, 40, 0, PF(P_D_C3), 192, PF(P_D_BN3_G),
            PF(P_D_BN3_B), bigA, ACT_SILU);
  k_reduce_mid<<<B * 192, 256, 0, s>>>(bigA, se_mean, 1600, 192, 0);
  gemm_lin(s, se_mean, PF(P_D_SE2_F1W), PF(P_D_SE2_F1B), se_hid, B, 48, 192,
           ACT_SILU);
  gemm_lin(s, se_hid, PF(P_D_SE2_F2W), PF(P_D_SE2_F2B), se_scale, B, 192, 48,
           ACT_SIGM);
  k_scale_channels<<<ew_grid((long)B * 1600 * 192), 256, 0, s>>>(
      bigA, se_scale, 1600, 192, (long)B * 1600 * 192);
  conv2d_bn(s, bigA, B, 192, 40, 40, 0, PF(P_D_C4), 192, PF(P_D_BN4_G),
            PF(P_D_BN4_B), bigB, ACT_SILU);
  k_reduce_mid<<<B * 192, 256, 0, s>>>(bigB, feat, 1600, 192, 0);
  k_layernorm<<<B, 256, 0, s>>>(feat, PF(P_DINORM_G), PF(P_DINORM_B),
                                dimg_state, 192);
  // ---- char branch --------------------------------------------------------
  conv1d_b(s, emb, B, 84, T, 5, PF(P_CC1_W), 192, PF(P_CC1_B), chA, ACT_SILU);
  conv1d_b(s, chA, B, 192, T, 3, PF(P_CC2_W), 192, PF(P_CC2_B), chB, ACT_SILU);
  conv1d_b(s, chB, B, 192, T, 3, PF(P_CC3_W), 192, PF(P_CC3_B), chA, ACT_SILU);
  k_reduce_mid<<<B * 192, 256, 0, s>>>(chA, feat, T, 192, 1);
  k_layernorm<<<B, 256, 0, s>>>(feat, PF(P_CHNORM_G), PF(P_CHNORM_B),
                                char_state, 192);
  // ---- word branch --------------------------------------------------------
  k_word_mean<<<B, 128, 0, s>>>(PF(P_WEMB), word_ids, feat, B, 64, 72);
  k_layernorm<<<B, 256, 0, s>>>(feat, PF(P_WNORM_G), PF(P_WNORM_B), word_state,
                                72);
  // ---- prompt branch ------------------------------------------------------
  gemm_lin(s, prompt, PF(P_PF1_W), PF(P_PF1_B), ptmp, B, 64, 16, ACT_SILU);
  gemm_lin(s, ptmp, PF(P_PF2_W), PF(P_PF2_B), ptmp2, B, 40, 64, ACT_NONE);
  k_layernorm<<<B, 256, 0, s>>>(ptmp2, PF(P_PRNORM_G), PF(P_PRNORM_B),
                                prompt_state, 40);
  // ---- fusion input -------------------------------------------------------
  k_copy2d<<<ew_grid(B * 320), 256, 0, s>>>(text_state, fin, B, 320, 925, 0);
  k_copy2d<<<ew_grid(B * 108), 256, 0, s>>>(image_state, fin, B, 108, 925,
                                            320);
  k_copy2d<<<ew_grid(B * 192), 256, 0, s>>>(char_state, fin, B, 192, 925, 428);
  k_copy2d<<<ew_grid(B * 72), 256, 0, s>>>(word_state, fin, B, 72, 925, 620);
  k_copy2d<<<ew_grid(B * 192), 256, 0, s>>>(dimg_state, fin, B, 192, 925, 692);
  k_copy2d<<<ew_grid(B * 40), 256, 0, s>>>(prompt_state, fin, B, 40, 925, 884);
  k_copy2d<<<ew_grid(B), 256, 0, s>>>(has_image, fin, B, 1, 925, 924);
  gemm_lin(s, fin, PF(P_FUSIN_W), PF(P_FUSIN_B), fused, B, F, 925, ACT_SILU);
  // ---- modality stack [B,7,F] --------------------------------------------
  gemm_lin(s, text_state, PF(P_TPROJ_W), PF(P_TPROJ_B), stack, B, F, 320,
           ACT_NONE, 7 * F, 0 * F);
  gemm_lin(s, image_state, PF(P_IMPROJ_W), PF(P_IMPROJ_B), stack, B, F, 108,
           ACT_NONE, 7 * F, 1 * F);
  gemm_lin(s, char_state, PF(P_CHPROJ_W), PF(P_CHPROJ_B), stack, B, F, 192,
           ACT_NONE, 7 * F, 2 * F);
  gemm_lin(s, word_state, PF(P_WPROJ_W), PF(P_WPROJ_B), stack, B, F, 72,
           ACT_NONE, 7 * F, 3 * F);
  gemm_lin(s, dimg_state, PF(P_DIMGPROJ_W), PF(P_DIMGPROJ_B), stack, B, F, 192,
           ACT_NONE, 7 * F, 4 * F);
  gemm_lin(s, prompt_state, PF(P_PRPROJ_W), PF(P_PRPROJ_B), stack, B, F, 40,
           ACT_NONE, 7 * F, 5 * F);
  gemm_lin(s, has_image, PF(P_HIPROJ_W), PF(P_HIPROJ_B), stack, B, F, 1,
           ACT_NONE, 7 * F, 6 * F);
  // ---- memory block -------------------------------------------------------
  gemm_lin(s, fused, PF(P_MEM_Q_W), PF(P_MEM_Q_B), attn, B, 12, F, ACT_NONE);
  k_softmax_rows<<<1, 64, 0, s>>>(attn, B, 12);
  gemm_lin(s, attn, PF(P_MEM_SLOTS), nullptr, memv, B, F, 12, ACT_NONE);
  k_copy2d<<<ew_grid(B * F), 256, 0, s>>>(fused, joined, B, F, 2 * F, 0);
  k_copy2d<<<ew_grid(B * F), 256, 0, s>>>(memv, joined, B, F, 2 * F, F);
  k_layernorm<<<B, 256, 0, s>>>(joined, PF(P_MEM_MLN_G), PF(P_MEM_MLN_B), jln,
                                2 * F);
  gemm_lin(s, jln, PF(P_MEM_MIX1_W), PF(P_MEM_MIX1_B), mixh, B, 2 * F, 2 * F,
           ACT_SILU);
  gemm_lin(s, mixh, PF(P_MEM_MIX2_W), PF(P_MEM_MIX2_B), upd, B, F, 2 * F,
           ACT_NONE);
  gemm_lin(s, joined, PF(P_MEM_GATE_W), PF(P_MEM_GATE_B), gatev, B, F, 2 * F,
           ACT_SIGM);
  gemm_lin(s, memv, PF(P_MEM_RES_W), PF(P_MEM_RES_B), resid, B, F, F,
           ACT_NONE);
  k_mem_combine<<<ew_grid(B * F), 256, 0, s>>>(fused, gatev, upd, resid,
                                               (long)B * F);
  // ---- 5-step context / expert loop ---------------------------------------
  const int cqB[5] = {P_CQ0_B, P_CQ1_B, P_CQ2_B, P_CQ3_B, P_CQ4_B};
  const int cqW[5] = {P_CQ0_W, P_CQ1_W, P_CQ2_W, P_CQ3_W, P_CQ4_W};
  const int erB[5] = {P_ER0_B, P_ER1_B, P_ER2_B, P_ER3_B, P_ER4_B};
  const int erW[5] = {P_ER0_W, P_ER1_W, P_ER2_W, P_ER3_W, P_ER4_W};
  const int exF1B[4] = {P_EX0_F1B, P_EX1_F1B, P_EX2_F1B, P_EX3_F1B};
  const int exF1W[4] = {P_EX0_F1W, P_EX1_F1W, P_EX2_F1W, P_EX3_F1W};
  const int exF2B[4] = {P_EX0_F2B, P_EX1_F2B, P_EX2_F2B, P_EX3_F2B};
  const int exF2W[4] = {P_EX0_F2W, P_EX1_F2W, P_EX2_F2W, P_EX3_F2W};
  const int exLNB[4] = {P_EX0_LNB, P_EX1_LNB, P_EX2_LNB, P_EX3_LNB};
  const int exLNG[4] = {P_EX0_LNG, P_EX1_LNG, P_EX2_LNG, P_EX3_LNG};
  for (int step = 0; step < 5; ++step) {
    gemm_lin(s, fused, PF(cqW[step]), PF(cqB[step]), cl, B, 7, F, ACT_NONE);
    k_softmax_rows<<<1, 64, 0, s>>>(cl, B, 7);
    k_loss_accum<<<1, 32, 0, s>>>(cl, B, 7, ctx_slot);
    k_topk_norm<<<1, 64, 0, s>>>(cl, cw, B, 7, 3);
    k_ctx_sum<<<B, 256, 0, s>>>(stack, cw, ctx, 7, F);
    k_axpby<<<ew_grid(B * F), 256, 0, s>>>(fused, ctx, tmpF, 1.f, 0.35f,
                                           (long)B * F);
    k_layernorm<<<B, 256, 0, s>>>(tmpF, PF(P_EINNORM_G), PF(P_EINNORM_B),
                                  routed, F);
    k_copy2d<<<ew_grid(B * F), 256, 0, s>>>(routed, rcat, B, F, 2 * F, 0);
    k_copy2d<<<ew_grid(B * F), 256, 0, s>>>(ctx, rcat, B, F, 2 * F, F);
    gemm_lin(s, rcat, PF(erW[step]), PF(erB[step]), rl, B, 4, 2 * F, ACT_NONE);
    k_softmax_rows<<<1, 64, 0, s>>>(rl, B, 4);
    k_loss_accum<<<1, 32, 0, s>>>(rl, B, 4, exp_slot);
    k_topk_norm<<<1, 64, 0, s>>>(rl, rw, B, 4, 2);
    k_axpby<<<ew_grid(B * F), 256, 0, s>>>(routed, ctx, ein, 1.f, 0.2f,
                                           (long)B * F);
    k_zero<<<ew_grid(B * F), 256, 0, s>>>(mixed, (long)B * F);
    for (int idx = 0; idx < 4; ++idx) {
      k_layernorm<<<B, 256, 0, s>>>(ein, PF(exLNG[idx]), PF(exLNB[idx]), eh,
                                    F);
      gemm_lin(s, eh, PF(exF1W[idx]), PF(exF1B[idx]), e1, B, 896, F, ACT_SILU);
      gemm_lin(s, e1, PF(exF2W[idx]), PF(exF2B[idx]), e2, B, F, 896, ACT_NONE);
      k_add<<<ew_grid(B * F), 256, 0, s>>>(ein, e2, eout, (long)B * F);
      k_row_scaled_add<<<ew_grid(B * F), 256, 0, s>>>(mixed, eout, rw, idx, 4,
                                                      F, (long)B * F);
    }
    k_copy2d<<<ew_grid(B * F), 256, 0, s>>>(ctx, ccat, B, F, 2 * F, 0);
    k_copy2d<<<ew_grid(B * F), 256, 0, s>>>(mixed, ccat, B, F, 2 * F, F);
    gemm_lin(s, ccat, PF(P_EMERGE_W), PF(P_EMERGE_B), delta, B, F, 2 * F,
             ACT_NONE);
    k_add<<<ew_grid(B * F), 256, 0, s>>>(fused, delta, fused, (long)B * F);
  }
  // ---- verifier + refiner + heads ----------------------------------------
  gemm_lin(s, fused, PF(P_PREINT_W), PF(P_PREINT_B), pre3, B, 128, F, ACT_NONE,
           260, 0);
  gemm_lin(s, fused, PF(P_PREDOM_W), PF(P_PREDOM_B), pre3, B, 32, F, ACT_NONE,
           260, 128);
  gemm_lin(s, fused, PF(P_PREVIS_W), PF(P_PREVIS_B), pre3, B, 100, F, ACT_NONE,
           260, 160);
  gemm_lin(s, pre3, PF(P_VP1_W), PF(P_VP1_B), vv1, B, 288, 260, ACT_SILU);
  gemm_lin(s, vv1, PF(P_VP2_W), PF(P_VP2_B), verifier, B, F, 288, ACT_NONE);
  k_copy2d<<<ew_grid(B * F), 256, 0, s>>>(fused, gcat, B, F, 2 * F, 0);
  k_copy2d<<<ew_grid(B * F), 256, 0, s>>>(verifier, gcat, B, F, 2 * F, F);
  gemm_lin(s, gcat, PF(P_VGATE_W), PF(P_VGATE_B), vg, B, F, 2 * F, ACT_SIGM);
  k_gate_combine<<<ew_grid(B * F), 256, 0, s>>>(fused, vg, verifier,
                                                (long)B * F);
  k_layernorm<<<B, 256, 0, s>>>(fused, PF(P_REF_LNG), PF(P_REF_LNB), eh, F);
  gemm_lin(s, eh, PF(P_REF_F1W), PF(P_REF_F1B), e1, B, 896, F, ACT_SILU);
  gemm_lin(s, e1, PF(P_REF_F2W), PF(P_REF_F2B), e2, B, F, 896, ACT_NONE);
  k_add<<<ew_grid(B * F), 256, 0, s>>>(fused, e2, refined, (long)B * F);
  gemm_lin(s, refined, PF(P_INTHEAD_W), PF(P_INTHEAD_B), out + 0, B, 128, F,
           ACT_NONE);
  gemm_lin(s, refined, PF(P_RESPHEAD_W), PF(P_RESPHEAD_B), out + 8192, B, 1024,
           F, ACT_NONE);
  gemm_lin(s, refined, PF(P_VISHEAD_W), PF(P_VISHEAD_B), out + 73728, B, 100,
           F, ACT_NONE);
  gemm_lin(s, refined, PF(P_DOMHEAD_W), PF(P_DOMHEAD_B), out + 80128, B, 32, F,
           ACT_NONE);
#undef PF
}
